// Head_30889404793528
// MI455X (gfx1250) — matrix operand order repaired and checked by reading, not compiled, not run on hardware
//
#include <hip/hip_runtime.h>
#include <hip/hip_bf16.h>

#define BB 4
#define TT 4096
#define CC 1024
#define HH 64

#define AS1 __attribute__((address_space(1)))
#define AS3 __attribute__((address_space(3)))

typedef __attribute__((ext_vector_type(16))) __bf16 v16bf;
typedef __attribute__((ext_vector_type(8)))  float  v8f;
typedef __attribute__((ext_vector_type(4)))  int    v4i;

union BF16x16 { v16bf v; unsigned int u[8]; };

__device__ __forceinline__ unsigned short bf_bits(float f) {
    return __builtin_bit_cast(unsigned short, (__bf16)f);   // v_cvt_pk_bf16_f32
}
__device__ __forceinline__ unsigned int pack_bf(float lo, float hi) {
    return (unsigned)bf_bits(lo) | ((unsigned)bf_bits(hi) << 16);
}
// A-operand k order used by this file (contiguous; B uses the same): VGPR j of lane (m=lane&15, half=lane>>4)
// holds K = {base, base+1}, base = 2j + 16*half.
__device__ __forceinline__ int a_kbase(int j, int half) {
    return 2 * j + 16 * half;
}

// 16-byte global -> LDS copy; async (ASYNCcnt) when the CDNA5 builtin exists.
__device__ __forceinline__ void async_cp16(const void* g, void* l) {
#if __has_builtin(__builtin_amdgcn_global_load_async_to_lds_b128)
    __builtin_amdgcn_global_load_async_to_lds_b128(
        (AS1 v4i*)const_cast<void*>(g), (AS3 v4i*)l, 0, 0);
#else
    *(uint4*)l = *(const uint4*)g;
#endif
}
__device__ __forceinline__ void async_join() {
#if __has_builtin(__builtin_amdgcn_s_wait_asynccnt)
    __builtin_amdgcn_s_wait_asynccnt(0);
#elif defined(__AMDGCN__)
    asm volatile("s_wait_asynccnt 0" ::: "memory");
#endif
}

// ---------------------------------------------------------------------------
// Kernel 0: W -> bf16, transposed to Wt[matrix][n][ch] so B fragments are
// contiguous in LDS later.  Tiny (384 KB out), bandwidth-trivial.
// ---------------------------------------------------------------------------
__global__ __launch_bounds__(256) void prep_w_kernel(
    const float* __restrict__ Wk, const float* __restrict__ Wq,
    const float* __restrict__ Wv, unsigned short* __restrict__ Wt)
{
    int i = blockIdx.x * 256 + threadIdx.x;          // pair index
    const int NPAIR = 3 * HH * (CC / 2);
    if (i >= NPAIR) return;
    int mz  = i / (HH * (CC / 2));
    int r   = i % (HH * (CC / 2));
    int n   = r / (CC / 2);
    int chp = (r % (CC / 2)) * 2;
    const float* W = (mz == 0) ? Wk : ((mz == 1) ? Wq : Wv);
    unsigned int v = pack_bf(W[(size_t)chp * HH + n], W[(size_t)(chp + 1) * HH + n]);
    ((unsigned int*)Wt)[(size_t)(mz * HH + n) * (CC / 2) + (r % (CC / 2))] = v;
}

// ---------------------------------------------------------------------------
// Kernel 1: fused q/k/v projection.  4 waves / 64 rows per WG; x chunk staged
// ONCE and reused for all 12 (matrix, n-tile) WMMAs; double-buffered async
// staging overlaps DMA of chunk kc+1 with WMMA compute on chunk kc.
// ---------------------------------------------------------------------------
__global__ __launch_bounds__(128) void proj_qkv_kernel(
    const float* __restrict__ x,
    const unsigned short* __restrict__ Wt,    // [3*64][1024] bf16, ch contiguous
    unsigned short* __restrict__ kq,
    unsigned short* __restrict__ qq,
    unsigned short* __restrict__ vq)
{
    __shared__ float          Xf[2][64][32];     // 2 x 8 KB  fp32 x chunks
    __shared__ unsigned short Wsh[2][192][32];   // 2 x 12 KB bf16 W chunks

    const int tid  = threadIdx.x;
    const int wave = tid >> 5;
    const int lane = tid & 31;
    const int half = lane >> 4;
    const int l15  = lane & 15;
    const long rowWG = (long)blockIdx.x * 64;
    const int  rloc  = wave * 16 + l15;

    auto stage = [&](int kc, int p) {
        // x chunk: 64 rows x 32 ch fp32 = 512 x 16B
#pragma unroll
        for (int i = 0; i < 4; ++i) {
            int idx = tid * 4 + i;
            int r   = idx >> 3;
            int seg = (idx & 7) * 4;          // float offset
            async_cp16(x + (rowWG + r) * CC + kc * 32 + seg, &Xf[p][r][seg]);
        }
        // W chunk: 192 cols x 32 ch bf16 = 768 x 16B
#pragma unroll
        for (int i = 0; i < 6; ++i) {
            int idx = tid * 6 + i;
            int c   = idx >> 2;
            int seg = (idx & 3) * 8;          // ushort offset
            async_cp16(Wt + (size_t)c * CC + kc * 32 + seg, &Wsh[p][c][seg]);
        }
    };

    v8f acc[12] = {};                         // [mz*4 + nt]

    stage(0, 0);
    for (int kc = 0; kc < CC / 32; ++kc) {
        const int p = kc & 1;
        async_join();                         // chunk kc landed in LDS
        __syncthreads();
        if (kc + 1 < CC / 32) stage(kc + 1, p ^ 1);   // overlap next DMA

        // A fragment once per chunk (contiguous fp32 pairs -> b64 ds loads)
        BF16x16 a;
#pragma unroll
        for (int j = 0; j < 8; ++j) {
            int kk = a_kbase(j, half);
            float2 f2 = *(const float2*)&Xf[p][rloc][kk];
            a.u[j] = pack_bf(f2.x, f2.y);
        }
        // 12 WMMAs reuse A; B fragments are two b128 ds loads each
#pragma unroll
        for (int mz = 0; mz < 3; ++mz) {
#pragma unroll
            for (int nt = 0; nt < 4; ++nt) {
                int col = mz * 64 + nt * 16 + l15;   // B: N on lanes
                BF16x16 bm;
                uint4 w0 = *(const uint4*)&Wsh[p][col][16 * half];
                uint4 w1 = *(const uint4*)&Wsh[p][col][16 * half + 8];
                bm.u[0] = w0.x; bm.u[1] = w0.y; bm.u[2] = w0.z; bm.u[3] = w0.w;
                bm.u[4] = w1.x; bm.u[5] = w1.y; bm.u[6] = w1.z; bm.u[7] = w1.w;
                acc[mz * 4 + nt] = __builtin_amdgcn_wmma_f32_16x16x32_bf16(
                    false, a.v, false, bm.v, (short)0, acc[mz * 4 + nt], false, false);
            }
        }
        __syncthreads();                      // buffer p free for kc+2
    }

    // D layout: lanes 0-15 -> (m=i, n=l15); lanes 16-31 -> (m=i+8).
#pragma unroll
    for (int mz = 0; mz < 3; ++mz) {
        unsigned short* outp = (mz == 0) ? kq : ((mz == 1) ? qq : vq);
        float os = (mz == 1) ? 0.03125f : 1.0f;      // fold 1/sqrt(C) into q
#pragma unroll
        for (int nt = 0; nt < 4; ++nt) {
#pragma unroll
            for (int i = 0; i < 8; ++i) {
                long r = rowWG + wave * 16 + i + 8 * half;
                outp[r * HH + nt * 16 + l15] = bf_bits(acc[mz * 4 + nt][i] * os);
            }
        }
    }
}

// ---------------------------------------------------------------------------
// Kernel 2: causal flash attention.  4 waves x 16 queries per WG; 32-key
// K/V tiles double-buffered via async-to-LDS; online softmax; P transposed
// through per-wave LDS scratch.
// ---------------------------------------------------------------------------
__global__ __launch_bounds__(128) void flash_attn_kernel(
    const unsigned short* __restrict__ qb,
    const unsigned short* __restrict__ kbuf,
    const unsigned short* __restrict__ vbuf,
    float* __restrict__ out)
{
    __shared__ unsigned short Ks[2][32][64];   // [buf][key][chan]
    __shared__ unsigned short Vs[2][32][64];
    __shared__ unsigned short Ps[4][16][32];   // per-wave P scratch

    const int tid  = threadIdx.x;
    const int wave = tid >> 5;
    const int lane = tid & 31;
    const int half = lane >> 4;
    const int l15  = lane & 15;

    const int b     = blockIdx.x / (TT / 64);
    const int qwg   = (blockIdx.x % (TT / 64)) * 64;
    const int qbase = qwg + wave * 16;
    const long brow = (long)b * TT;

    auto stage = [&](int kblk, int p) {
        const unsigned short* kg = kbuf + (brow + kblk * 32) * HH;
        const unsigned short* vg = vbuf + (brow + kblk * 32) * HH;
#pragma unroll
        for (int i = 0; i < 2; ++i) {
            int idx = tid * 2 + i;
            int r   = idx >> 3;
            int seg = (idx & 7) * 8;           // ushort offset
            async_cp16(kg + r * HH + seg, &Ks[p][r][seg]);
            async_cp16(vg + r * HH + seg, &Vs[p][r][seg]);
        }
    };

    // Q (pre-scaled) in A layout, channel chunks 0-31 / 32-63
    BF16x16 aq0, aq1;
    {
        const unsigned short* qrow = qb + (brow + qbase + l15) * HH;
#pragma unroll
        for (int j = 0; j < 8; ++j) {
            int kk = a_kbase(j, half);
            aq0.u[j] = *(const unsigned int*)(qrow + kk);
            aq1.u[j] = *(const unsigned int*)(qrow + 32 + kk);
        }
    }

    float mrow[8], lrow[8];
    v8f acc[4] = {{}, {}, {}, {}};
#pragma unroll
    for (int i = 0; i < 8; ++i) { mrow[i] = -__builtin_inff(); lrow[i] = 0.f; }

    const int nkb = (qwg + 63) / 32 + 1;
    stage(0, 0);
    for (int kblk = 0; kblk < nkb; ++kblk) {
        const int p = kblk & 1;
        async_join();                          // tile kblk landed in LDS
        __syncthreads();
        if (kblk + 1 < nkb) stage(kblk + 1, p ^ 1);   // overlap next DMA
        if (kblk + 2 < nkb)                    // pull tile kblk+2 toward L2
            __builtin_prefetch(kbuf + (brow + (kblk + 2) * 32 + (tid >> 2)) * HH, 0, 1);

        if (kblk * 32 <= qbase + 15) {         // wave-uniform causal skip
            // ---- S = q @ k^T : two 16-key tiles, K-dim 64 = 2x32 ----
            v8f s[2];
#pragma unroll
            for (int j = 0; j < 2; ++j) {
                BF16x16 bk0, bk1;
                int key = j * 16 + l15;
#pragma unroll
                for (int jj = 0; jj < 8; ++jj) {
                    int ch = half * 16 + 2 * jj;
                    bk0.u[jj] = *(const unsigned int*)&Ks[p][key][ch];
                    bk1.u[jj] = *(const unsigned int*)&Ks[p][key][32 + ch];
                }
                v8f z = {};
                z = __builtin_amdgcn_wmma_f32_16x16x32_bf16(
                    false, aq0.v, false, bk0.v, (short)0, z, false, false);
                z = __builtin_amdgcn_wmma_f32_16x16x32_bf16(
                    false, aq1.v, false, bk1.v, (short)0, z, false, false);
                s[j] = z;
            }

            // ---- causal mask + online softmax ----
            const int gk0 = kblk * 32 + l15;
            const int gk1 = gk0 + 16;
            float p0[8], p1[8];
#pragma unroll
            for (int i = 0; i < 8; ++i) {
                int gq = qbase + i + 8 * half;
                float s0 = (gk0 <= gq) ? s[0][i] : -__builtin_inff();
                float s1 = (gk1 <= gq) ? s[1][i] : -__builtin_inff();
                float rm = fmaxf(s0, s1);
                rm = fmaxf(rm, __shfl_xor(rm, 1, 32));
                rm = fmaxf(rm, __shfl_xor(rm, 2, 32));
                rm = fmaxf(rm, __shfl_xor(rm, 4, 32));
                rm = fmaxf(rm, __shfl_xor(rm, 8, 32));
                float mn   = fmaxf(mrow[i], rm);
                float corr = __expf(mrow[i] - mn);
                float e0   = __expf(s0 - mn);
                float e1   = __expf(s1 - mn);
                float rs = e0 + e1;
                rs += __shfl_xor(rs, 1, 32);
                rs += __shfl_xor(rs, 2, 32);
                rs += __shfl_xor(rs, 4, 32);
                rs += __shfl_xor(rs, 8, 32);
                lrow[i] = lrow[i] * corr + rs;
                mrow[i] = mn;
                p0[i] = e0;
                p1[i] = e1;
#pragma unroll
                for (int h = 0; h < 4; ++h) acc[h][i] *= corr;
            }

            // ---- transpose P (D layout -> A layout) through LDS ----
#pragma unroll
            for (int i = 0; i < 8; ++i) {
                int m = i + 8 * half;
                Ps[wave][m][l15]      = bf_bits(p0[i]);
                Ps[wave][m][16 + l15] = bf_bits(p1[i]);
            }
            asm volatile("s_wait_dscnt 0" ::: "memory");
            BF16x16 pa;
#pragma unroll
            for (int j = 0; j < 8; ++j) {
                int kk = a_kbase(j, half);
                pa.u[j] = *(const unsigned int*)&Ps[wave][l15][kk];
            }

            // ---- acc += P @ V ----
#pragma unroll
            for (int h = 0; h < 4; ++h) {
                BF16x16 bv;
                int hd = h * 16 + l15;
#pragma unroll
                for (int jj = 0; jj < 8; ++jj) {
                    int k0 = half * 16 + 2 * jj;
                    unsigned lo = Vs[p][k0][hd];
                    unsigned hi = Vs[p][k0 + 1][hd];
                    bv.u[jj] = lo | (hi << 16);
                }
                acc[h] = __builtin_amdgcn_wmma_f32_16x16x32_bf16(
                    false, pa.v, false, bv.v, (short)0, acc[h], false, false);
            }
        }
        __syncthreads();                       // buffer p free for kblk+2
    }

    // normalize + store fp32
#pragma unroll
    for (int i = 0; i < 8; ++i) {
        float inv = 1.0f / lrow[i];
        int gq = qbase + i + 8 * half;
        float* orow = out + (brow + gq) * HH;
#pragma unroll
        for (int h = 0; h < 4; ++h)
            orow[h * 16 + l15] = acc[h][i] * inv;
    }
}

// ---------------------------------------------------------------------------
extern "C" void kernel_launch(void* const* d_in, const int* in_sizes, int n_in,
                              void* d_out, int out_size, void* d_ws, size_t ws_size,
                              hipStream_t stream) {
    (void)in_sizes; (void)n_in; (void)out_size; (void)ws_size;
    const float* x  = (const float*)d_in[0];
    const float* Wk = (const float*)d_in[1];
    const float* Wq = (const float*)d_in[2];
    const float* Wv = (const float*)d_in[3];

    const size_t nTok = (size_t)BB * TT * HH;           // 1,048,576
    const size_t nWt  = (size_t)3 * HH * CC;            // 196,608
    unsigned short* Wt = (unsigned short*)d_ws;         // 384 KB
    unsigned short* kq = Wt + nWt;
    unsigned short* qq = kq + nTok;
    unsigned short* vq = qq + nTok;

    prep_w_kernel<<<dim3((3 * HH * (CC / 2) + 255) / 256), 256, 0, stream>>>(
        Wk, Wq, Wv, Wt);

    proj_qkv_kernel<<<dim3(BB * TT / 64), 128, 0, stream>>>(x, Wt, kq, qq, vq);

    flash_attn_kernel<<<dim3(BB * (TT / 64)), 128, 0, stream>>>(
        qq, kq, vq, (float*)d_out);
}
